// E3Conv_79207786873044
// MI455X (gfx1250) — matrix operand order repaired and checked by reading, not compiled, not run on hardware
//
#include <hip/hip_runtime.h>

// ---------------------------------------------------------------------------
// E(3)-equivariant graph conv for MI455X (gfx1250, wave32, WMMA).
// Heavy tensor-product contraction mapped to v_wmma_f32_16x16x32_bf16:
//   tmp[e, si*128+g] = sum_f m[e,f] * tpw[f,si,g]   (GEMM M=E, K=128, N=1152)
//   agg[dst[e], g]  += sum_si sh[e,si] * tmp[e, si*128+g]
// B (tpw, bf16) lives entirely in LDS (288 KB of the WGP's 320 KB).
// ---------------------------------------------------------------------------

typedef __attribute__((ext_vector_type(16))) __bf16          v16bf;
typedef __attribute__((ext_vector_type(8)))  float           v8f;
typedef __attribute__((ext_vector_type(16))) unsigned short  v16u;
typedef __attribute__((ext_vector_type(8)))  unsigned short  v8u;

constexpr int kS  = 9;     // spherical harmonic channels
constexpr int kF  = 128;   // feature dim
constexpr int kA  = 16;    // edge attr dim
constexpr int kHR = 64;    // radial hidden
constexpr int kTileE = 16; // edges per WMMA tile (M)
constexpr float kInvSqrtDeg = 0.17677669529663687f; // 1/sqrt(32)

static __device__ __forceinline__ unsigned short f32_to_bf16(float f) {
  unsigned int u = __float_as_uint(f);
  unsigned int r = 0x7FFFu + ((u >> 16) & 1u);   // round-to-nearest-even
  return (unsigned short)((u + r) >> 16);
}

// ---------------------------------------------------------------------------
// Per-edge geometry: spherical harmonics (9) + edge attributes (16).
// ---------------------------------------------------------------------------
__global__ void edge_prep_kernel(const float* __restrict__ pos,
                                 const int* __restrict__ ei,        // [2,E]
                                 const int* __restrict__ bond_mask, // [E]
                                 const float* __restrict__ bond_embed, // [2,8]
                                 const int* __restrict__ cutoff_p,
                                 float* __restrict__ edge_sh,   // [E,9]
                                 float* __restrict__ edge_attr, // [E,16]
                                 int E_) {
  int e = blockIdx.x * blockDim.x + threadIdx.x;
  if (e >= E_) return;
  int s = ei[e], d = ei[E_ + e];
  float vx = pos[s*3+0] - pos[d*3+0];
  float vy = pos[s*3+1] - pos[d*3+1];
  float vz = pos[s*3+2] - pos[d*3+2];
  float rr = sqrtf(vx*vx + vy*vy + vz*vz);
  float inv = 1.0f / fmaxf(rr, 1e-12f);
  float x = vx*inv, y = vy*inv, z = vz*inv;
  const float c1  = 1.7320508075688772f;   // sqrt(3)
  const float c2  = 3.872983346207417f;    // sqrt(15)
  const float c2m = 1.118033988749895f;    // sqrt(5)/2
  float* sh = edge_sh + (size_t)e * kS;
  sh[0] = 1.0f;
  sh[1] = c1 * y;
  sh[2] = c1 * z;
  sh[3] = c1 * x;
  sh[4] = c2 * x * y;
  sh[5] = c2 * y * z;
  sh[6] = c2m * (3.0f*z*z - 1.0f);
  sh[7] = c2 * x * z;
  sh[8] = (c2 * 0.5f) * (x*x - y*y);

  float cutoff = (float)(*cutoff_p);
  float step = cutoff / 9.0f;              // linspace(0,c,10) spacing
  float* ea = edge_attr + (size_t)e * kA;
  int bm = bond_mask[e];
  #pragma unroll
  for (int i = 0; i < 8; ++i) ea[i] = bond_embed[bm*8 + i];
  #pragma unroll
  for (int i = 0; i < 8; ++i) {
    float v = step * (float)(i + 1);
    float diff = (rr - v) / step;
    ea[8 + i] = __expf(-diff*diff) / 1.12f;
  }
}

// ---------------------------------------------------------------------------
// x init: gather atom embeddings.
// ---------------------------------------------------------------------------
__global__ void init_x_kernel(const float* __restrict__ atom_embed,
                              const int* __restrict__ atom_types,
                              float* __restrict__ x, int total) {
  int i = blockIdx.x * blockDim.x + threadIdx.x;
  if (i >= total) return;
  int n = i >> 7, f = i & 127;
  x[i] = atom_embed[atom_types[n]*kF + f];
}

// ---------------------------------------------------------------------------
// Radial MLP + node scaling, fused: m[e,f] = (silu(ea@W1)@W2)[f]
//                                            * x[src,f] * (1 + c_noise[src]*gamma[f])
// Output m is bf16 (WMMA A operand precision).
// ---------------------------------------------------------------------------
__global__ __launch_bounds__(256)
void radial_m_kernel(const float* __restrict__ edge_attr, // [E,16]
                     const int* __restrict__ src,         // [E]
                     const float* __restrict__ x,         // [N,128]
                     const float* __restrict__ c_noise,   // [N]
                     const float* __restrict__ gamma,     // [128]
                     const float* __restrict__ rw1,       // [16,64]
                     const float* __restrict__ rw2,       // [64,128]
                     unsigned short* __restrict__ m,      // [E,128] bf16
                     int E_) {
  __shared__ float w1[kA * kHR];
  __shared__ float w2[kHR * kF];
  for (int i = threadIdx.x; i < kA*kHR; i += 256) w1[i] = rw1[i];
  for (int i = threadIdx.x; i < kHR*kF; i += 256) w2[i] = rw2[i];
  __syncthreads();

  int e = blockIdx.x * 256 + threadIdx.x;
  if (e >= E_) return;

  float a[kA];
  #pragma unroll
  for (int i = 0; i < kA; ++i) a[i] = edge_attr[(size_t)e*kA + i];

  float h[kHR];
  #pragma unroll
  for (int j = 0; j < kHR; ++j) {
    float s = 0.0f;
    #pragma unroll
    for (int i = 0; i < kA; ++i) s += a[i] * w1[i*kHR + j];
    h[j] = s / (1.0f + __expf(-s));       // silu
  }

  int sn = src[e];
  float cn = c_noise[sn];
  const float* xr = x + (size_t)sn * kF;
  unsigned short* mr = m + (size_t)e * kF;
  for (int f = 0; f < kF; ++f) {
    float r = 0.0f;
    #pragma unroll
    for (int j = 0; j < kHR; ++j) r += h[j] * w2[j*kF + f];
    float scale = 1.0f + cn * gamma[f];
    mr[f] = f32_to_bf16(r * xr[f] * scale);
  }
}

// ---------------------------------------------------------------------------
// WMMA tensor-product GEMM + sh-weighted si-reduction + scatter-add.
// Block = 256 threads = 8 waves. Wave w owns output columns g = 16w..16w+15.
// B (tpw as bf16, layout [si][g][k], k contiguous) fully LDS-resident.
// ---------------------------------------------------------------------------
__global__ __launch_bounds__(256)
void conv_gemm_kernel(const unsigned short* __restrict__ m,  // [E,128] bf16
                      const float* __restrict__ edge_sh,     // [E,9]
                      const int* __restrict__ dst,           // [E]
                      const float* __restrict__ tpw,         // [128,9,128] f32
                      float* __restrict__ agg,               // [N,128] f32
                      int E_) {
  __shared__ alignas(32) unsigned short bLds[kS * kF * kF]; // 288 KB
  __shared__ alignas(32) unsigned short mLds[kTileE * kF];  // 4 KB
  __shared__ float shLds[kTileE * kS];
  __shared__ int dstLds[kTileE];

  const int tid = threadIdx.x;

  // Stage B once per block: bLds[si*16384 + g*128 + k] = bf16(tpw[k, si, g]).
  for (int idx = tid; idx < kS*kF*kF; idx += 256) {
    int si  = idx >> 14;
    int rem = idx & 16383;
    int g   = rem >> 7;
    int k   = rem & 127;
    bLds[idx] = f32_to_bf16(tpw[(size_t)k*(kS*kF) + si*kF + g]);
  }
  __syncthreads();

  const int lane = tid & 31;
  const int gt   = tid >> 5;          // wave id == g-tile
  const int nloc = lane & 15;         // A row / B column within tile
  const int hi   = lane >> 4;         // lane half selects K sub-range
  const int g    = gt*16 + nloc;

  const int numTiles = E_ / kTileE;
  for (int tile = blockIdx.x; tile < numTiles; tile += gridDim.x) {
    const int e0 = tile * kTileE;

    // Stage m tile / sh tile / dst indices.
    {
      const unsigned short* msrc = m + (size_t)e0 * kF;
      for (int idx = tid*8; idx < kTileE*kF; idx += 256*8)
        *(v8u*)&mLds[idx] = *(const v8u*)&msrc[idx];
      if (tid < kTileE*kS) shLds[tid] = edge_sh[(size_t)e0*kS + tid];
      if (tid < kTileE)    dstLds[tid] = dst[e0 + tid];
    }
    // Prefetch next tile's m stream (global_prefetch_b8).
    if (tile + gridDim.x < numTiles)
      __builtin_prefetch(m + (size_t)(tile + gridDim.x)*kTileE*kF + tid*8, 0, 1);
    __syncthreads();

    v8f zero = {0.f,0.f,0.f,0.f,0.f,0.f,0.f,0.f};
    v8f acc[kS];
    #pragma unroll
    for (int si = 0; si < kS; ++si) acc[si] = zero;

    #pragma unroll
    for (int kc = 0; kc < 4; ++kc) {
      // A fragment: lane half 0 -> K 0..15, lane half 1 -> K 16..31
      // (contiguous, matching B), offset by kc*32 in f.
      const int baseA = kc*32 + hi*16;
      v16u au = *(const v16u*)&mLds[nloc*kF + baseA];
      v16bf aFrag = __builtin_bit_cast(v16bf, au);

      // B fragments: lane half 0 -> K 0..15, half 1 -> K 16..31 (contiguous).
      const int baseB = kc*32 + hi*16;
      #pragma unroll
      for (int si = 0; si < kS; ++si) {
        v16u bu = *(const v16u*)&bLds[si*(kF*kF) + g*kF + baseB];
        v16bf bFrag = __builtin_bit_cast(v16bf, bu);
        acc[si] = __builtin_amdgcn_wmma_f32_16x16x32_bf16(
            false, aFrag, false, bFrag, (short)0, acc[si], false, false);
      }
    }

    // Epilogue: C layout -> lane half selects M 0-7 / 8-15; column = g.
    const int rowBase = hi * 8;
    #pragma unroll
    for (int j = 0; j < 8; ++j) {
      const int row = rowBase + j;
      float v = 0.0f;
      #pragma unroll
      for (int si = 0; si < kS; ++si) v += shLds[row*kS + si] * acc[si][j];
      atomicAdd(&agg[(size_t)dstLds[row]*kF + g], v);
    }
    __syncthreads();
  }
}

// ---------------------------------------------------------------------------
// Layer-0: x = agg / sqrt(avg_deg).
// ---------------------------------------------------------------------------
__global__ void scale0_kernel(const float* __restrict__ agg,
                              float* __restrict__ x, int total) {
  int i = blockIdx.x * blockDim.x + threadIdx.x;
  if (i >= total) return;
  x[i] = agg[i] * kInvSqrtDeg;
}

// ---------------------------------------------------------------------------
// Gated residual: x = gate*x + (1-gate)*agg/sqrt(deg).
// ---------------------------------------------------------------------------
__global__ void gate_kernel(float* __restrict__ x,
                            const float* __restrict__ agg,
                            const float* __restrict__ c_noise,
                            const float* __restrict__ gw,
                            const float* __restrict__ gb, int total) {
  int i = blockIdx.x * blockDim.x + threadIdx.x;
  if (i >= total) return;
  int n = i >> 7, f = i & 127;
  float t = c_noise[n] * gw[f] + gb[f];
  float gate = 1.0f / (1.0f + __expf(-t));
  x[i] = gate * x[i] + (1.0f - gate) * (agg[i] * kInvSqrtDeg);
}

// ---------------------------------------------------------------------------
// Output projection: out = x @ w_out * gain.
// ---------------------------------------------------------------------------
__global__ void out_kernel(const float* __restrict__ x,
                           const float* __restrict__ w_out,  // [128,3]
                           const float* __restrict__ gain_p,
                           float* __restrict__ out, int N_) {
  int n = blockIdx.x * blockDim.x + threadIdx.x;
  if (n >= N_) return;
  float gain = *gain_p;
  const float* xr = x + (size_t)n * kF;
  float a0 = 0.f, a1 = 0.f, a2 = 0.f;
  #pragma unroll 8
  for (int f = 0; f < kF; ++f) {
    float xv = xr[f];
    a0 += xv * w_out[f*3 + 0];
    a1 += xv * w_out[f*3 + 1];
    a2 += xv * w_out[f*3 + 2];
  }
  out[n*3 + 0] = a0 * gain;
  out[n*3 + 1] = a1 * gain;
  out[n*3 + 2] = a2 * gain;
}

// ---------------------------------------------------------------------------
extern "C" void kernel_launch(void* const* d_in, const int* in_sizes, int n_in,
                              void* d_out, int out_size, void* d_ws, size_t ws_size,
                              hipStream_t stream) {
  const float* pos        = (const float*)d_in[0];
  const int*   ei         = (const int*)  d_in[1];
  const int*   bond_mask  = (const int*)  d_in[2];
  const int*   atom_types = (const int*)  d_in[3];
  const float* c_noise    = (const float*)d_in[4];
  const int*   cutoff_p   = (const int*)  d_in[5];
  const float* atom_embed = (const float*)d_in[6];
  const float* bond_embed = (const float*)d_in[7];
  const float* rw1        = (const float*)d_in[8];   // [5,16,64]
  const float* rw2        = (const float*)d_in[9];   // [5,64,128]
  const float* tpw        = (const float*)d_in[10];  // [5,128,9,128]
  const float* gammas     = (const float*)d_in[11];  // [5,128]
  const float* gate_w     = (const float*)d_in[12];  // [4,128]
  const float* gate_b     = (const float*)d_in[13];  // [4,128]
  const float* w_out      = (const float*)d_in[14];  // [128,3]
  const float* gain_p     = (const float*)d_in[15];

  const int E_ = in_sizes[1] / 2;
  const int N_ = in_sizes[4];
  const int* src = ei;
  const int* dst = ei + E_;

  // Workspace carve-out (all 256B aligned).
  char* ws = (char*)d_ws;
  size_t off = 0;
  auto carve = [&](size_t bytes) {
    char* p = ws + off;
    off += (bytes + 255) & ~(size_t)255;
    return p;
  };
  float*          edge_sh   = (float*)         carve((size_t)E_ * kS * 4);
  float*          edge_attr = (float*)         carve((size_t)E_ * kA * 4);
  unsigned short* mbuf      = (unsigned short*)carve((size_t)E_ * kF * 2);
  float*          xbuf      = (float*)         carve((size_t)N_ * kF * 4);
  float*          aggbuf    = (float*)         carve((size_t)N_ * kF * 4);
  (void)ws_size;

  const int nodeTotal = N_ * kF;

  edge_prep_kernel<<<(E_ + 255)/256, 256, 0, stream>>>(
      pos, ei, bond_mask, bond_embed, cutoff_p, edge_sh, edge_attr, E_);
  init_x_kernel<<<(nodeTotal + 255)/256, 256, 0, stream>>>(
      atom_embed, atom_types, xbuf, nodeTotal);

  for (int l = 0; l < 5; ++l) {
    radial_m_kernel<<<(E_ + 255)/256, 256, 0, stream>>>(
        edge_attr, src, xbuf, c_noise,
        gammas + (size_t)l * kF,
        rw1 + (size_t)l * kA * kHR,
        rw2 + (size_t)l * kHR * kF,
        mbuf, E_);
    hipMemsetAsync(aggbuf, 0, (size_t)nodeTotal * 4, stream);
    conv_gemm_kernel<<<1024, 256, 0, stream>>>(
        mbuf, edge_sh, dst, tpw + (size_t)l * kF * kS * kF, aggbuf, E_);
    if (l == 0) {
      scale0_kernel<<<(nodeTotal + 255)/256, 256, 0, stream>>>(
          aggbuf, xbuf, nodeTotal);
    } else {
      gate_kernel<<<(nodeTotal + 255)/256, 256, 0, stream>>>(
          xbuf, aggbuf, c_noise,
          gate_w + (size_t)(l-1) * kF,
          gate_b + (size_t)(l-1) * kF, nodeTotal);
    }
  }

  out_kernel<<<(N_ + 255)/256, 256, 0, stream>>>(
      xbuf, w_out, gain_p, (float*)d_out, N_);
}
